// SelfAttention_26242250178598
// MI455X (gfx1250) — matrix operand order repaired and checked by reading, not compiled, not run on hardware
//
#include <hip/hip_runtime.h>
#include <hip/hip_bf16.h>
#include <stdint.h>

// Problem constants (match reference)
#define E_DIM    1024
#define THREE_E  3072
#define NUM_HEADS 16
#define HEAD_DIM  64
#define SEQ      2048
#define BATCH    4
#define M_TOTAL  (BATCH * SEQ)   // 8192 rows

typedef __attribute__((ext_vector_type(16))) __bf16 v16bf;
typedef __attribute__((ext_vector_type(2)))  __bf16 v2bf;
typedef __attribute__((ext_vector_type(8)))  float  v8f;
typedef __attribute__((ext_vector_type(4)))  int    v4i;

union FragAB { v16bf v; uint4 q[2]; };

// ---------------- bf16 conversion helpers ----------------
// Use the native __bf16 conversion so ISel picks the hardware cvt op
// (v_cvt_pk_bf16_f32 / scalar cvt) instead of a manual rounding sequence.
__device__ __forceinline__ unsigned short f2bf(float f) {
    __bf16 h = (__bf16)f;
    return __builtin_bit_cast(unsigned short, h);
}
__device__ __forceinline__ unsigned int pk_bf16(float a, float b) {
    v2bf r = { (__bf16)a, (__bf16)b };
    return __builtin_bit_cast(unsigned int, r);
}

// ---------------- async global->LDS copy (gfx1250 ASYNCcnt path) ----------------
#if __has_builtin(__builtin_amdgcn_global_load_async_to_lds_b128)
#define HAVE_ASYNC_LDS 1
__device__ __forceinline__ void async_copy_b128(const void* g, void* l) {
    __builtin_amdgcn_global_load_async_to_lds_b128(
        (__attribute__((address_space(1))) v4i*)g,
        (__attribute__((address_space(3))) v4i*)l, 0, 0);
}
__device__ __forceinline__ void wait_async0() {
#if __has_builtin(__builtin_amdgcn_s_wait_asynccnt)
    __builtin_amdgcn_s_wait_asynccnt(0);
#else
    asm volatile("s_wait_asynccnt 0x0" ::: "memory");
#endif
}
#else
#define HAVE_ASYNC_LDS 0
__device__ __forceinline__ void async_copy_b128(const void* g, void* l) {
    *(uint4*)l = *(const uint4*)g;
}
__device__ __forceinline__ void wait_async0() {}
#endif

// ---------------- WMMA fragment loaders (bf16 16x16x32, wave32) ----------------
// A (16x32): lane m = lane&15, half = lane>>4.
//   VGPRs 0..3 hold K = k0 + half*8 + 0..7   (16 contiguous bytes)
//   VGPRs 4..7 hold K = k0 + 16 + half*8 + 0..7
__device__ __forceinline__ void load_afrag(FragAB& f, const unsigned short* lds,
                                           int row0, int ld, int k0, int lane) {
    int m = lane & 15, half = lane >> 4;
    const unsigned short* p = lds + (size_t)(row0 + m) * ld + k0 + half * 8;
    f.q[0] = *(const uint4*)p;
    f.q[1] = *(const uint4*)(p + 16);
}
// B (32x16) loaded from LDS tile stored TRANSPOSED as Bt[n][k]:
//   lane n = lane&15, half = lane>>4; VGPRs 0..3 hold K = k0 + half*8 + 0..7,
//   VGPRs 4..7 hold K = k0 + 16 + half*8 + 0..7 -> 2x b128.
__device__ __forceinline__ void load_btfrag(FragAB& f, const unsigned short* ldsT,
                                            int n0, int ld, int k0, int lane) {
    int n = lane & 15, half = lane >> 4;
    const unsigned short* p = ldsT + (size_t)(n0 + n) * ld + k0 + half * 8;
    f.q[0] = *(const uint4*)p;
    f.q[1] = *(const uint4*)(p + 16);
}

// =====================================================================
// Kernel 1: QKV projection  qkv = bf16(x @ W1 + b1)   (8192 x 3072)
// 256 threads = 8 waves in 2(M) x 4(N); tile 128x128, BK=32.
// =====================================================================
__global__ __launch_bounds__(256)
void qkv_gemm(const float* __restrict__ x, const float* __restrict__ W1,
              const float* __restrict__ b1, unsigned short* __restrict__ qkv) {
    __shared__ __align__(16) unsigned short aT[128 * 40];  // [m][k], padded
    __shared__ __align__(16) unsigned short bT[128 * 40];  // [n][k] (transposed W tile)
    const int t = threadIdx.x;
    const int lane = t & 31, w = t >> 5;
    const int half = lane >> 4, nl = lane & 15;
    const int wm = w >> 2, wn = w & 3;
    const int m0 = blockIdx.x * 128;
    const int n0 = blockIdx.y * 128;

    v8f acc[4][2];
    for (int i = 0; i < 4; i++)
        for (int j = 0; j < 2; j++)
            for (int e = 0; e < 8; e++) acc[i][j][e] = 0.f;

    for (int kk = 0; kk < E_DIM; kk += 32) {
        // x tile: 128x32 fp32 -> bf16 into aT[m][k] (float4 along k, b64 LDS store)
        for (int i = 0; i < 4; i++) {
            int linear = t + i * 256;
            int row = linear >> 3;
            int cg  = (linear & 7) * 4;
            const float4 f = *(const float4*)(x + (size_t)(m0 + row) * E_DIM + kk + cg);
            uint2 pk; pk.x = pk_bf16(f.x, f.y); pk.y = pk_bf16(f.z, f.w);
            *(uint2*)&aT[row * 40 + cg] = pk;
        }
        // W1 tile: 32x128 fp32 -> bf16 transposed into bT[n][k].
        // 4 scalar loads ALONG k (lane-coalesced along n) so the transposed
        // store is one aligned ds_store_b64 of 4 packed bf16.
        for (int i = 0; i < 4; i++) {
            int linear = t + i * 256;      // 0..1023
            int n  = linear & 127;
            int kg = linear >> 7;          // 0..7 -> k = kg*4..kg*4+3
            const float* wp = W1 + (size_t)(kk + kg * 4) * THREE_E + n0 + n;
            float f0 = wp[0];
            float f1 = wp[THREE_E];
            float f2 = wp[2 * THREE_E];
            float f3 = wp[3 * THREE_E];
            uint2 pk; pk.x = pk_bf16(f0, f1); pk.y = pk_bf16(f2, f3);
            *(uint2*)&bT[n * 40 + kg * 4] = pk;
        }
        __syncthreads();

        FragAB a[4], b[2];
        for (int mt = 0; mt < 4; mt++) load_afrag(a[mt], aT, wm * 64 + mt * 16, 40, 0, lane);
        for (int nt = 0; nt < 2; nt++) load_btfrag(b[nt], bT, wn * 32 + nt * 16, 40, 0, lane);
        for (int mt = 0; mt < 4; mt++)
            for (int nt = 0; nt < 2; nt++)
                acc[mt][nt] = __builtin_amdgcn_wmma_f32_16x16x32_bf16(
                    false, a[mt].v, false, b[nt].v, (short)0, acc[mt][nt], false, false);
        __syncthreads();
    }

    // epilogue: + bias, store bf16 (C layout: m = r + 8*half, n = nl)
    for (int nt = 0; nt < 2; nt++) {
        int n = n0 + wn * 32 + nt * 16 + nl;
        float bias = b1[n];
        for (int mt = 0; mt < 4; mt++) {
            int mbase = m0 + wm * 64 + mt * 16 + 8 * half;
            for (int r = 0; r < 8; r++)
                qkv[(size_t)(mbase + r) * THREE_E + n] = f2bf(acc[mt][nt][r] + bias);
        }
    }
}

// =====================================================================
// Kernel 2: causal flash attention over bf16 QKV.
// Grid: (B*H, S/64). 128 threads = 4 waves, each wave owns 16 query rows.
// Per head h in the 3072-wide QKV row: Q at h*192, K at h*192+64, V at +128.
// =====================================================================
__global__ __launch_bounds__(128)
void flash_attn(const unsigned short* __restrict__ qkv,
                unsigned short* __restrict__ attn) {
    __shared__ __align__(16) unsigned short kT[64 * 72];      // K tile [key][d] (== Bt for Q@K^T)
    __shared__ __align__(16) unsigned short vT[64 * 72];      // V^T tile [d][key] (== Bt for P@V)
    __shared__ __align__(16) unsigned short pT[4 * 16 * 72];  // per-wave P tile [16][64]
    const int t = threadIdx.x;
    const int lane = t & 31, wq = t >> 5;
    const int half = lane >> 4, nl = lane & 15;
    const int b = blockIdx.x >> 4, h = blockIdx.x & 15;
    const int qb = blockIdx.y;
    const int q0 = qb * 64;
    const int qcol = h * 192, kcol = h * 192 + 64, vcol = h * 192 + 128;

    // Q fragments: 16 rows x 64 d, kept in registers for the whole pass
    FragAB qa0, qa1;
    {
        int qrow = b * SEQ + q0 + wq * 16 + nl;
        const unsigned short* qp = qkv + (size_t)qrow * THREE_E + qcol;
        qa0.q[0] = *(const uint4*)(qp + half * 8);
        qa0.q[1] = *(const uint4*)(qp + 16 + half * 8);
        qa1.q[0] = *(const uint4*)(qp + 32 + half * 8);
        qa1.q[1] = *(const uint4*)(qp + 48 + half * 8);
    }

    float mrun[8], lrun[8];
    v8f o[4];
    for (int r = 0; r < 8; r++) { mrun[r] = -3.0e38f; lrun[r] = 0.f; }
    for (int i = 0; i < 4; i++)
        for (int e = 0; e < 8; e++) o[i][e] = 0.f;

    unsigned short* myP = pT + wq * 16 * 72;

    for (int kb = 0; kb <= qb; kb++) {
        // cooperative load: K tile via async global->LDS (pure bf16 copy),
        // V tile transposed through VGPRs.
        for (int i = 0; i < 4; i++) {
            int linear = t + i * 128;
            int key = linear >> 3;
            int dg  = (linear & 7) * 8;
            const unsigned short* kp = qkv + (size_t)(b * SEQ + kb * 64 + key) * THREE_E;
            async_copy_b128(kp + kcol + dg, &kT[key * 72 + dg]);
            uint4 vv = *(const uint4*)(kp + vcol + dg);
            const unsigned short* vs = (const unsigned short*)&vv;
            for (int j = 0; j < 8; j++) vT[(dg + j) * 72 + key] = vs[j];
        }
        if (kb < qb) {  // prefetch next K tile -> global_prefetch_b8
            const unsigned short* np =
                qkv + (size_t)(b * SEQ + (kb + 1) * 64 + (t >> 1)) * THREE_E + kcol + (t & 1) * 32;
            __builtin_prefetch(np, 0, 1);
        }
        wait_async0();
        __syncthreads();

        // S = Q @ K^T  (4 x 16-key tiles, K-depth 64 = 2 WMMA steps)
        v8f sacc[4];
        for (int nt = 0; nt < 4; nt++) {
            v8f z; for (int e = 0; e < 8; e++) z[e] = 0.f;
            FragAB bk0, bk1;
            load_btfrag(bk0, kT, nt * 16, 72, 0,  lane);
            load_btfrag(bk1, kT, nt * 16, 72, 32, lane);
            z = __builtin_amdgcn_wmma_f32_16x16x32_bf16(false, qa0.v, false, bk0.v, (short)0, z, false, false);
            z = __builtin_amdgcn_wmma_f32_16x16x32_bf16(false, qa1.v, false, bk1.v, (short)0, z, false, false);
            sacc[nt] = z;
        }

        // online softmax; each (half, r) pair = one query row, stats replicated
        // across its 16 lanes; reductions stay within the 16-lane half.
        for (int r = 0; r < 8; r++) {
            int qg = q0 + wq * 16 + r + 8 * half;
            float bm = -3.0e38f;
            for (int nt = 0; nt < 4; nt++) {
                int c = kb * 64 + nt * 16 + nl;
                float s = sacc[nt][r] * 0.125f;        // 1/sqrt(64)
                if (c > qg) s = -1.0e10f;               // causal mask (matches ref)
                sacc[nt][r] = s;
                bm = fmaxf(bm, s);
            }
            bm = fmaxf(bm, __shfl_xor(bm, 1, 32));
            bm = fmaxf(bm, __shfl_xor(bm, 2, 32));
            bm = fmaxf(bm, __shfl_xor(bm, 4, 32));
            bm = fmaxf(bm, __shfl_xor(bm, 8, 32));
            float mnew = fmaxf(mrun[r], bm);
            float corr = __expf(mrun[r] - mnew);
            mrun[r] = mnew;
            float ls = 0.f;
            for (int nt = 0; nt < 4; nt++) {
                float p = __expf(sacc[nt][r] - mnew);
                ls += p;
                myP[(r + 8 * half) * 72 + nt * 16 + nl] = f2bf(p);
            }
            ls += __shfl_xor(ls, 1, 32);
            ls += __shfl_xor(ls, 2, 32);
            ls += __shfl_xor(ls, 4, 32);
            ls += __shfl_xor(ls, 8, 32);
            lrun[r] = lrun[r] * corr + ls;
            for (int nt2 = 0; nt2 < 4; nt2++) o[nt2][r] *= corr;
        }
        // P was written C-layout, now re-read A-layout: drain LDS stores.
        asm volatile("s_wait_dscnt 0" ::: "memory");

        // O += P @ V
        FragAB pa0, pa1;
        load_afrag(pa0, myP, 0, 72, 0,  lane);
        load_afrag(pa1, myP, 0, 72, 32, lane);
        for (int nt2 = 0; nt2 < 4; nt2++) {
            FragAB bv0, bv1;
            load_btfrag(bv0, vT, nt2 * 16, 72, 0,  lane);
            load_btfrag(bv1, vT, nt2 * 16, 72, 32, lane);
            o[nt2] = __builtin_amdgcn_wmma_f32_16x16x32_bf16(false, pa0.v, false, bv0.v, (short)0, o[nt2], false, false);
            o[nt2] = __builtin_amdgcn_wmma_f32_16x16x32_bf16(false, pa1.v, false, bv1.v, (short)0, o[nt2], false, false);
        }
        __syncthreads();
    }

    // normalize and store attn (B,S,E) as bf16; E index = h*64 + d
    for (int r = 0; r < 8; r++) {
        int row = b * SEQ + q0 + wq * 16 + r + 8 * half;
        float inv = 1.0f / lrun[r];
        for (int nt2 = 0; nt2 < 4; nt2++)
            attn[(size_t)row * E_DIM + h * 64 + nt2 * 16 + nl] = f2bf(o[nt2][r] * inv);
    }
}

// =====================================================================
// Kernel 3: out = attn @ W2 + b2 (fp32 out). Same tiling as kernel 1.
// A tile is already bf16: copied with async global->LDS (ASYNCcnt path).
// =====================================================================
__global__ __launch_bounds__(256)
void out_gemm(const unsigned short* __restrict__ attn, const float* __restrict__ W2,
              const float* __restrict__ b2, float* __restrict__ out) {
    __shared__ __align__(16) unsigned short aT[128 * 40];
    __shared__ __align__(16) unsigned short bT[128 * 40];
    const int t = threadIdx.x;
    const int lane = t & 31, w = t >> 5;
    const int half = lane >> 4, nl = lane & 15;
    const int wm = w >> 2, wn = w & 3;
    const int m0 = blockIdx.x * 128;
    const int n0 = blockIdx.y * 128;

    v8f acc[4][2];
    for (int i = 0; i < 4; i++)
        for (int j = 0; j < 2; j++)
            for (int e = 0; e < 8; e++) acc[i][j][e] = 0.f;

    for (int kk = 0; kk < E_DIM; kk += 32) {
        // attn tile: straight bf16 b128 copies via async-to-LDS.
        for (int i = 0; i < 2; i++) {
            int linear = t + i * 256;
            int row = linear >> 2;
            int cg  = (linear & 3) * 8;
            async_copy_b128(attn + (size_t)(m0 + row) * E_DIM + kk + cg,
                            &aT[row * 40 + cg]);
        }
        // W2 tile: 32x128 fp32 -> bf16 transposed (loads along k, b64 store)
        for (int i = 0; i < 4; i++) {
            int linear = t + i * 256;
            int n  = linear & 127;
            int kg = linear >> 7;
            const float* wp = W2 + (size_t)(kk + kg * 4) * E_DIM + n0 + n;
            float f0 = wp[0];
            float f1 = wp[E_DIM];
            float f2 = wp[2 * E_DIM];
            float f3 = wp[3 * E_DIM];
            uint2 pk; pk.x = pk_bf16(f0, f1); pk.y = pk_bf16(f2, f3);
            *(uint2*)&bT[n * 40 + kg * 4] = pk;
        }
        wait_async0();
        __syncthreads();

        FragAB a[4], b[2];
        for (int mt = 0; mt < 4; mt++) load_afrag(a[mt], aT, wm * 64 + mt * 16, 40, 0, lane);
        for (int nt = 0; nt < 2; nt++) load_btfrag(b[nt], bT, wn * 32 + nt * 16, 40, 0, lane);
        for (int mt = 0; mt < 4; mt++)
            for (int nt = 0; nt < 2; nt++)
                acc[mt][nt] = __builtin_amdgcn_wmma_f32_16x16x32_bf16(
                    false, a[mt].v, false, b[nt].v, (short)0, acc[mt][nt], false, false);
        __syncthreads();
    }

    for (int nt = 0; nt < 2; nt++) {
        int n = n0 + wn * 32 + nt * 16 + nl;
        float bias = b2[n];
        for (int mt = 0; mt < 4; mt++) {
            int mbase = m0 + wm * 64 + mt * 16 + 8 * half;
            for (int r = 0; r < 8; r++)
                out[(size_t)(mbase + r) * E_DIM + n] = acc[mt][nt][r] + bias;
        }
    }
}

extern "C" void kernel_launch(void* const* d_in, const int* in_sizes, int n_in,
                              void* d_out, int out_size, void* d_ws, size_t ws_size,
                              hipStream_t stream) {
    const float* x  = (const float*)d_in[0];
    const float* W1 = (const float*)d_in[1];
    const float* b1 = (const float*)d_in[2];
    const float* W2 = (const float*)d_in[3];
    const float* b2 = (const float*)d_in[4];

    // workspace: QKV bf16 (8192*3072 = 50.3MB) then attn bf16 (8192*1024 = 16.8MB)
    unsigned short* qkv  = (unsigned short*)d_ws;
    unsigned short* attn = qkv + (size_t)M_TOTAL * THREE_E;
    float* out = (float*)d_out;

    qkv_gemm<<<dim3(M_TOTAL / 128, THREE_E / 128), 256, 0, stream>>>(x, W1, b1, qkv);
    flash_attn<<<dim3(BATCH * NUM_HEADS, SEQ / 64), 128, 0, stream>>>(qkv, attn);
    out_gemm<<<dim3(M_TOTAL / 128, E_DIM / 128), 256, 0, stream>>>(attn, W2, b2, out);
}
